// ContactPredictionHead_17136919511475
// MI455X (gfx1250) — compile-verified, matrix operand order checked by reading, not run on hardware
//
#include <hip/hip_runtime.h>
#include <math.h>

// Problem constants (from the reference): B=2, L=2048, D=1536, T=2
#define L_DIM 2048
#define D_DIM 1536
#define B_DIM 2

// Tiling
#define BM 128        // i-rows per workgroup
#define BN 128        // j-cols per workgroup
#define BK 32         // K slab (matches WMMA 16x16x32)
#define LDS_K 40      // padded K stride in halves (80B = 20 banks)

#define NTB  (L_DIM / BM)            // 16 tile-blocks per dim
#define NTRI (NTB * (NTB + 1) / 2)   // 136 upper-triangular block pairs

#define TILE_HALVES (3 * BM * LDS_K)              // halves per pipeline stage (30720 B)

// Transpose staging: per-wave 16 (j) x 64 (i) float2, padded row stride 66
#define STG_STRIDE   66
#define STG_PER_WAVE (16 * STG_STRIDE)            // float2 elements per wave
#define SMEM_BYTES   (8 * STG_PER_WAVE * 8)       // 67584 B (> 2 stages * 30720 B)

typedef __attribute__((ext_vector_type(16))) _Float16 v16h;
typedef __attribute__((ext_vector_type(8)))  float    v8f;
typedef __attribute__((ext_vector_type(4)))  _Float16 h4;
typedef __attribute__((ext_vector_type(2)))  float    f32x2;
typedef __attribute__((ext_vector_type(4)))  float    f32x4;

union FragU {
    v16h   v;
    float4 q[2];
};

__global__ __launch_bounds__(256) void contact_head_wmma_kernel(
    const float* __restrict__ H,     // (B, L, D) f32
    const float* __restrict__ W,     // (T, 2D) f32 ; Wp[t] = W[t*2D .. t*2D+D)
    const float* __restrict__ bias,  // (T,) f32
    float* __restrict__ out)         // (B, L, L, T) f32
{
    __shared__ __align__(16) unsigned char smem[SMEM_BYTES];
    f32x2* stg = (f32x2*)smem;                    // transpose staging (aliased)

    // ---- decode upper-triangular block pair: p = by*(by+1)/2 + bx, bx <= by
    const int p  = (int)blockIdx.x;
    int by = (int)((sqrtf(8.0f * (float)p + 1.0f) - 1.0f) * 0.5f);
    while ((by + 1) * (by + 2) / 2 <= p) ++by;
    while (by * (by + 1) / 2 > p) --by;
    const int bx = p - by * (by + 1) / 2;
    const int i0 = bx * BM;
    const int j0 = by * BN;
    const int bz = (int)blockIdx.z;

    const float* Hb = H + (size_t)bz * L_DIM * D_DIM;
    const float* w0 = W;                  // Wp, t=0
    const float* w1 = W + 2 * D_DIM;      // Wp, t=1

    const int tid   = (int)threadIdx.x;
    // loader mapping: 256 threads cover 128 rows x 32 cols in float4 chunks
    const int lrow  = tid >> 1;           // 0..127
    const int lcolb = (tid & 1) << 4;     // 0 or 16
    // wave mapping: 8 waves in 2 (m) x 4 (n); each wave owns 64x32 of output
    const int lane = tid & 31;
    const int wv   = tid >> 5;
    const int wm   = (wv & 1) * 64;       // row offset inside block
    const int wn   = (wv >> 1) * 32;      // col offset inside block
    const int lr   = lane & 15;
    const int khi  = (lane >> 4) << 3;    // 0 (lanes 0-15) or 8 (lanes 16-31)

    // accumulators: [m-tile][n-tile][t]
    v8f acc[4][2][2] = {};

    // ---- register prefetch buffers (software pipeline, stage: global -> regs)
    float4 ra[4], rb[4], rw0[4], rw1[4];
    auto load_slab = [&](int kk) {
        #pragma unroll
        for (int q = 0; q < 4; ++q) {
            const int k = kk + lcolb + q * 4;
            ra[q]  = *(const float4*)&Hb[(size_t)(i0 + lrow) * D_DIM + k];
            rb[q]  = *(const float4*)&Hb[(size_t)(j0 + lrow) * D_DIM + k];
            rw0[q] = *(const float4*)&w0[k];
            rw1[q] = *(const float4*)&w1[k];
        }
    };

    // ---- stage: regs -> LDS (cvt f32->f16, A pre-scaled by w0/w1)
    auto produce = [&](int cur) {
        _Float16* sA0 = (_Float16*)smem + cur * TILE_HALVES;
        _Float16* sA1 = sA0 + BM * LDS_K;
        _Float16* sB  = sA1 + BM * LDS_K;
        #pragma unroll
        for (int q = 0; q < 4; ++q) {
            h4 a0v, a1v, bbv;
            a0v.x = (_Float16)(ra[q].x * rw0[q].x); a0v.y = (_Float16)(ra[q].y * rw0[q].y);
            a0v.z = (_Float16)(ra[q].z * rw0[q].z); a0v.w = (_Float16)(ra[q].w * rw0[q].w);
            a1v.x = (_Float16)(ra[q].x * rw1[q].x); a1v.y = (_Float16)(ra[q].y * rw1[q].y);
            a1v.z = (_Float16)(ra[q].z * rw1[q].z); a1v.w = (_Float16)(ra[q].w * rw1[q].w);
            bbv.x = (_Float16)rb[q].x; bbv.y = (_Float16)rb[q].y;
            bbv.z = (_Float16)rb[q].z; bbv.w = (_Float16)rb[q].w;
            const int soff = lrow * LDS_K + lcolb + q * 4;
            *(h4*)&sA0[soff] = a0v;
            *(h4*)&sA1[soff] = a1v;
            *(h4*)&sB [soff] = bbv;
        }
    };

    // ---- stage: LDS -> frags -> WMMA
    auto compute = [&](int cur) {
        const _Float16* sA0 = (const _Float16*)smem + cur * TILE_HALVES;
        const _Float16* sA1 = sA0 + BM * LDS_K;
        const _Float16* sB  = sA1 + BM * LDS_K;

        // B fragments (raw H_j): lane l<16: row l, K {0..7,16..23};
        // lane l>=16: row l-16, K {8..15,24..31}
        FragU fb[2];
        #pragma unroll
        for (int jn = 0; jn < 2; ++jn) {
            const _Float16* pB = &sB[(wn + jn * 16 + lr) * LDS_K + khi];
            fb[jn].q[0] = *(const float4*)(pB);
            fb[jn].q[1] = *(const float4*)(pB + 16);
        }
        #pragma unroll
        for (int im = 0; im < 4; ++im) {
            FragU fa0, fa1;
            const _Float16* p0 = &sA0[(wm + im * 16 + lr) * LDS_K + khi];
            const _Float16* p1 = &sA1[(wm + im * 16 + lr) * LDS_K + khi];
            fa0.q[0] = *(const float4*)(p0);
            fa0.q[1] = *(const float4*)(p0 + 16);
            fa1.q[0] = *(const float4*)(p1);
            fa1.q[1] = *(const float4*)(p1 + 16);

            #pragma unroll
            for (int jn = 0; jn < 2; ++jn) {
                acc[im][jn][0] = __builtin_amdgcn_wmma_f32_16x16x32_f16(
                    false, fa0.v, false, fb[jn].v, (short)0, acc[im][jn][0], false, false);
                acc[im][jn][1] = __builtin_amdgcn_wmma_f32_16x16x32_f16(
                    false, fa1.v, false, fb[jn].v, (short)0, acc[im][jn][1], false, false);
            }
        }
    };

    // ---- pipelined K loop: one barrier per iteration (double-buffered LDS)
    load_slab(0);
    produce(0);
    __syncthreads();
    int cur = 0;
    #pragma unroll 1
    for (int kk = 0; kk < D_DIM; kk += BK) {
        const bool more = (kk + BK) < D_DIM;
        if (more) load_slab(kk + BK);   // global loads in flight under WMMAs
        compute(cur);                   // consume stage `cur`
        if (more) produce(cur ^ 1);     // fill stage `cur^1` (waits on loads)
        __syncthreads();                // publish cur^1 + retire reads of cur
        cur ^= 1;
    }

    const float bias0 = bias[0];
    const float bias1 = bias[1];

    // ---- direct epilogue: out[b, i, j, t] as coalesced float2 {t0, t1}
    //      non-temporal: output is write-once, keep H resident in L2
    #pragma unroll
    for (int im = 0; im < 4; ++im) {
        #pragma unroll
        for (int jn = 0; jn < 2; ++jn) {
            #pragma unroll
            for (int r = 0; r < 8; ++r) {
                // C/D layout: VGPR r, lane l -> M = r + 8*(l>>4), N = l&15
                const int gi = i0 + wm + im * 16 + khi + r;
                const int gj = j0 + wn + jn * 16 + lr;
                const size_t off = (((size_t)(bz * L_DIM + gi) * L_DIM) + gj) * 2;
                f32x2 v;
                v.x = acc[im][jn][0][r] + bias0;
                v.y = acc[im][jn][1][r] + bias1;
                __builtin_nontemporal_store(v, (f32x2*)&out[off]);
            }
        }
    }

    // ---- mirror epilogue for off-diagonal blocks: out[b, j, i, t] = same tile,
    //      transposed through LDS so the global stores stay coalesced.
    if (bx != by) {
        f32x2* wstg = stg + wv * STG_PER_WAVE;    // this wave's staging region
        #pragma unroll
        for (int jn = 0; jn < 2; ++jn) {
            __syncthreads();   // staging aliases tile LDS / previous pass
            #pragma unroll
            for (int im = 0; im < 4; ++im) {
                #pragma unroll
                for (int r = 0; r < 8; ++r) {
                    const int iw = im * 16 + khi + r;  // 0..63 (wave-local i)
                    const int jw = lr;                 // 0..15 (wave-local j)
                    f32x2 v;
                    v.x = acc[im][jn][0][r] + bias0;
                    v.y = acc[im][jn][1][r] + bias1;
                    wstg[jw * STG_STRIDE + iw] = v;
                }
            }
            __syncthreads();
            // read back transposed: one j-row per iteration, 2 float2 per lane
            #pragma unroll
            for (int it = 0; it < 16; ++it) {
                const f32x4 v = *(const f32x4*)&wstg[it * STG_STRIDE + lane * 2];
                const int gj = j0 + wn + jn * 16 + it;
                const int gi = i0 + wm + lane * 2;
                const size_t off = (((size_t)(bz * L_DIM + gj) * L_DIM) + gi) * 2;
                __builtin_nontemporal_store(v, (f32x4*)&out[off]);
            }
        }
    }
}

extern "C" void kernel_launch(void* const* d_in, const int* in_sizes, int n_in,
                              void* d_out, int out_size, void* d_ws, size_t ws_size,
                              hipStream_t stream) {
    const float* H    = (const float*)d_in[0];  // hidden_states (B,L,D) f32
    const float* W    = (const float*)d_in[1];  // W (T, 2D) f32
    const float* bias = (const float*)d_in[2];  // b (T,) f32
    float* out        = (float*)d_out;          // (B,L,L,T) f32

    dim3 grid(NTRI, 1, B_DIM);
    dim3 block(256);
    contact_head_wmma_kernel<<<grid, block, 0, stream>>>(H, W, bias, out);
}
